// GeniePathLayer_29248727286045
// MI455X (gfx1250) — compile-verified, matrix operand order checked by reading, not run on hardware
//
#include <hip/hip_runtime.h>
#include <math.h>

// ---------------------------------------------------------------------------
// Types for CDNA5 WMMA (gfx1250, wave32)
// ---------------------------------------------------------------------------
typedef __attribute__((ext_vector_type(16))) __bf16 v16bf;
typedef __attribute__((ext_vector_type(8)))  float  v8f;
typedef __attribute__((ext_vector_type(4)))  unsigned int u32x4;

union FragU { v16bf v; u32x4 q[2]; };

__device__ __forceinline__ __bf16 f2bf(float f) {
  unsigned u = __float_as_uint(f);
  u += 0x7FFFu + ((u >> 16) & 1u);          // round-to-nearest-even
  unsigned short hs = (unsigned short)(u >> 16);
  return __builtin_bit_cast(__bf16, hs);
}

// Load a 16-element bf16 fragment: elements 0..7 from p0, 8..15 from p1.
// Matches the ISA 16-bit A/B fragment layout: for lane (half = lane>>4,
// r = lane&15), element i holds K = (i>>3)*16 + half*8 + (i&7).
__device__ __forceinline__ v16bf load_frag(const __bf16* p0, const __bf16* p1) {
  FragU f;
  f.q[0] = *(const u32x4*)p0;
  f.q[1] = *(const u32x4*)p1;
  return f.v;
}

__device__ __forceinline__ float sigmoidf_(float x) {
  return 1.0f / (1.0f + __expf(-x));
}

__device__ __forceinline__ void atomicMaxF(float* addr, float val) {
  if (val >= 0.0f) atomicMax((int*)addr, __float_as_int(val));
  else             atomicMin((unsigned int*)addr, __float_as_uint(val));
}

#define NEG_SLOPE 0.2f

// ---------------------------------------------------------------------------
// 0) Pack weights: Wgat_t[n*64+k] = bf16(W_gat[k*64+n])   (64x64, K-contig)
//    Wc_t[j*128+k] = bf16(k<64 ? W_ih[j*64+k] : W_hh[j*64+k-64])  (256x128)
// ---------------------------------------------------------------------------
__global__ void prep_weights_kernel(const float* __restrict__ Wgat,
                                    const float* __restrict__ Wih,
                                    const float* __restrict__ Whh,
                                    __bf16* __restrict__ wgt,
                                    __bf16* __restrict__ wct) {
  int tid = blockIdx.x * blockDim.x + threadIdx.x;
  if (tid < 64 * 64) {
    int n = tid >> 6, k = tid & 63;
    wgt[n * 64 + k] = f2bf(Wgat[k * 64 + n]);
  } else if (tid < 64 * 64 + 256 * 128) {
    int j2 = tid - 64 * 64;
    int row = j2 >> 7, k = j2 & 127;
    float v = (k < 64) ? Wih[row * 64 + k] : Whh[row * 64 + (k - 64)];
    wct[row * 128 + k] = f2bf(v);
  }
}

// ---------------------------------------------------------------------------
// 1) Init: agg = 0, m = -inf, z = 0
// ---------------------------------------------------------------------------
__global__ void init_kernel(float* __restrict__ agg, float* __restrict__ m,
                            float* __restrict__ z, int N) {
  size_t idx = (size_t)blockIdx.x * blockDim.x + threadIdx.x;
  if (idx < (size_t)N * 64) {
    agg[idx] = 0.0f;
    if (idx < (size_t)N) {
      m[idx] = -__builtin_inff();
      z[idx] = 0.0f;
    }
  }
}

// ---------------------------------------------------------------------------
// 2) Fused GAT projection + attention logits.
//    One wave per 16-node row block:
//      xs[16,64] = bf16(x[16,64]) @ W_gat      (8 WMMAs, A loaded once)
//      a_src[n]  = xs[n].att_src, a_dst[n] = xs[n].att_dst  (in-register,
//                  shfl_xor reduction within each 16-lane half)
// ---------------------------------------------------------------------------
__global__ void gemm_xs_attn_kernel(const float* __restrict__ x,
                                    const __bf16* __restrict__ Bt, // [64n x 64k]
                                    const float* __restrict__ att_s,
                                    const float* __restrict__ att_d,
                                    float* __restrict__ xs,
                                    float* __restrict__ a_src,
                                    float* __restrict__ a_dst,
                                    int Mtiles) {
  int wave = (int)((blockIdx.x * blockDim.x + threadIdx.x) >> 5);
  int lane = threadIdx.x & 31;
  if (wave >= Mtiles) return;
  int hf = lane >> 4, r = lane & 15;

  // Build A fragments from f32 x, converting to bf16 in-register.
  const float* arow = x + (size_t)(wave * 16 + r) * 64;
  v16bf afr[2];
#pragma unroll
  for (int ks = 0; ks < 2; ++ks) {
    int k0 = ks * 32 + hf * 8;
    float t0[8], t1[8];
    *(float4*)(t0)     = *(const float4*)(arow + k0);
    *(float4*)(t0 + 4) = *(const float4*)(arow + k0 + 4);
    *(float4*)(t1)     = *(const float4*)(arow + k0 + 16);
    *(float4*)(t1 + 4) = *(const float4*)(arow + k0 + 20);
    v16bf a;
#pragma unroll
    for (int i = 0; i < 8; ++i) { a[i] = f2bf(t0[i]); a[i + 8] = f2bf(t1[i]); }
    afr[ks] = a;
  }

  // 4 column tiles, K = 64 = 2 k-steps of 32.
  v8f acc[4] = {};
#pragma unroll
  for (int nt = 0; nt < 4; ++nt) {
    const __bf16* brow = Bt + (size_t)(nt * 16 + r) * 64;
#pragma unroll
    for (int ks = 0; ks < 2; ++ks) {
      int k0 = ks * 32 + hf * 8;
      acc[nt] = __builtin_amdgcn_wmma_f32_16x16x32_bf16(
          false, afr[ks], false, load_frag(brow + k0, brow + k0 + 16),
          (short)0, acc[nt], false, false);
    }
  }

  // Store xs tile (f32).
#pragma unroll
  for (int nt = 0; nt < 4; ++nt) {
    float* crow = xs + (size_t)(wave * 16) * 64 + nt * 16 + r;
#pragma unroll
    for (int i = 0; i < 8; ++i)
      crow[(size_t)(i + 8 * hf) * 64] = acc[nt][i];
  }

  // Attention logits: lane holds D[m = i+8*hf][n = r+16*nt].
  float ps[8], pd[8];
#pragma unroll
  for (int i = 0; i < 8; ++i) { ps[i] = 0.0f; pd[i] = 0.0f; }
#pragma unroll
  for (int nt = 0; nt < 4; ++nt) {
    float as_ = att_s[nt * 16 + r];
    float ad_ = att_d[nt * 16 + r];
#pragma unroll
    for (int i = 0; i < 8; ++i) {
      ps[i] += acc[nt][i] * as_;
      pd[i] += acc[nt][i] * ad_;
    }
  }
  // Reduce over the 16 lanes of each half (xor offsets 1..8 stay in-half).
#pragma unroll
  for (int off = 1; off < 16; off <<= 1) {
#pragma unroll
    for (int i = 0; i < 8; ++i) {
      ps[i] += __shfl_xor(ps[i], off);
      pd[i] += __shfl_xor(pd[i], off);
    }
  }
  if (r == 0) {
#pragma unroll
    for (int i = 0; i < 8; ++i) {
      int node = wave * 16 + i + 8 * hf;
      a_src[node] = ps[i];
      a_dst[node] = pd[i];
    }
  }
}

// ---------------------------------------------------------------------------
// 3) Edge pass 1: e = leaky_relu(a_src[s] + a_dst[d]); store; atomic max m[d]
// ---------------------------------------------------------------------------
__global__ void edge1_kernel(const int* __restrict__ src,
                             const int* __restrict__ dst,
                             const float* __restrict__ a_src,
                             const float* __restrict__ a_dst,
                             float* __restrict__ e_buf,
                             float* __restrict__ m, int E) {
  int e = blockIdx.x * blockDim.x + threadIdx.x;
  if (e >= E) return;
  int s = src[e], d = dst[e];
  float v = a_src[s] + a_dst[d];
  v = (v > 0.0f) ? v : NEG_SLOPE * v;
  e_buf[e] = v;
  atomicMaxF(&m[d], v);
}

// ---------------------------------------------------------------------------
// 4) Edge pass 2: p = exp(e - m[d]); store in place; hardware f32 atomic add
// ---------------------------------------------------------------------------
__global__ void edge2_kernel(const int* __restrict__ dst,
                             float* __restrict__ e_buf,
                             const float* __restrict__ m,
                             float* __restrict__ z, int E) {
  int e = blockIdx.x * blockDim.x + threadIdx.x;
  if (e >= E) return;
  int d = dst[e];
  float p = __expf(e_buf[e] - m[d]);
  e_buf[e] = p;
  unsafeAtomicAdd(&z[d], p);   // global_atomic_add_f32
}

// ---------------------------------------------------------------------------
// 5) Edge pass 3: agg[d][h] += xs[s][h] * (p / (z[d]+eps)).
//    Block (64,4): lane = channel -> coalesced 256B gather + scatter.
// ---------------------------------------------------------------------------
__global__ void edge3_kernel(const int* __restrict__ src,
                             const int* __restrict__ dst,
                             const float* __restrict__ p_buf,
                             const float* __restrict__ z,
                             const float* __restrict__ xs,
                             float* __restrict__ agg, int E) {
  int e = blockIdx.x * blockDim.y + threadIdx.y;
  if (e >= E) return;
  int h = threadIdx.x;
  int s = src[e], d = dst[e];
  float alpha = p_buf[e] / (z[d] + 1e-16f);
  unsafeAtomicAdd(&agg[(size_t)d * 64 + h], xs[(size_t)s * 64 + h] * alpha);
}

// ---------------------------------------------------------------------------
// 6) Build fused LSTM A-operand: xh[n, 0:64]  = bf16(tanh(agg + bias))
//                                xh[n,64:128] = bf16(h0)
// ---------------------------------------------------------------------------
__global__ void build_xh_kernel(const float* __restrict__ agg,
                                const float* __restrict__ bias,
                                const float* __restrict__ h0,
                                __bf16* __restrict__ xh, int N) {
  size_t idx = (size_t)blockIdx.x * blockDim.x + threadIdx.x;
  if (idx >= (size_t)N * 64) return;
  size_t n = idx >> 6;
  int h = (int)(idx & 63);
  xh[n * 128 + h]      = f2bf(tanhf(agg[idx] + bias[h]));
  xh[n * 128 + 64 + h] = f2bf(h0[idx]);
}

// ---------------------------------------------------------------------------
// 7) Fused gates GEMM + LSTM epilogue.
//    gates[N,256] = xh[N,128] @ Wc_t^T ; one wave per 16-node row block.
//    For each h-subtile t (columns t*16+r), accumulate the i/f/g/o tiles
//    (4 WMMAs each over K=128) and apply the LSTM cell in registers,
//    writing h1, h1, c1 straight to d_out.
// ---------------------------------------------------------------------------
__global__ void gemm_lstm_kernel(const __bf16* __restrict__ A,   // [N x 128]
                                 const __bf16* __restrict__ Wct, // [256 x 128]
                                 const float* __restrict__ c0,
                                 float* __restrict__ out,
                                 int Mtiles, size_t NH) {
  int wave = (int)((blockIdx.x * blockDim.x + threadIdx.x) >> 5);
  int lane = threadIdx.x & 31;
  if (wave >= Mtiles) return;
  int hf = lane >> 4, r = lane & 15;

  const __bf16* arow = A + (size_t)(wave * 16 + r) * 128;
  v16bf afr[4];
#pragma unroll
  for (int ks = 0; ks < 4; ++ks) {
    int k0 = ks * 32 + hf * 8;
    afr[ks] = load_frag(arow + k0, arow + k0 + 16);
  }

#pragma unroll
  for (int t = 0; t < 4; ++t) {
    const __bf16* bi = Wct + (size_t)(0   + t * 16 + r) * 128;
    const __bf16* bf = Wct + (size_t)(64  + t * 16 + r) * 128;
    const __bf16* bg = Wct + (size_t)(128 + t * 16 + r) * 128;
    const __bf16* bo = Wct + (size_t)(192 + t * 16 + r) * 128;
    v8f ai = {}, afv = {}, ag = {}, ao = {};
#pragma unroll
    for (int ks = 0; ks < 4; ++ks) {
      int k0 = ks * 32 + hf * 8;
      ai  = __builtin_amdgcn_wmma_f32_16x16x32_bf16(
          false, afr[ks], false, load_frag(bi + k0, bi + k0 + 16), (short)0, ai,  false, false);
      afv = __builtin_amdgcn_wmma_f32_16x16x32_bf16(
          false, afr[ks], false, load_frag(bf + k0, bf + k0 + 16), (short)0, afv, false, false);
      ag  = __builtin_amdgcn_wmma_f32_16x16x32_bf16(
          false, afr[ks], false, load_frag(bg + k0, bg + k0 + 16), (short)0, ag,  false, false);
      ao  = __builtin_amdgcn_wmma_f32_16x16x32_bf16(
          false, afr[ks], false, load_frag(bo + k0, bo + k0 + 16), (short)0, ao,  false, false);
    }
    int h = t * 16 + r;
#pragma unroll
    for (int i = 0; i < 8; ++i) {
      int node = wave * 16 + i + 8 * hf;
      size_t idx = (size_t)node * 64 + h;
      float c0v = c0[idx];
      float c1 = sigmoidf_(afv[i]) * c0v + sigmoidf_(ai[i]) * tanhf(ag[i]);
      float h1 = sigmoidf_(ao[i]) * tanhf(c1);
      out[idx]          = h1;
      out[NH + idx]     = h1;
      out[2 * NH + idx] = c1;
    }
  }
}

// ---------------------------------------------------------------------------
// Host launcher
// ---------------------------------------------------------------------------
extern "C" void kernel_launch(void* const* d_in, const int* in_sizes, int n_in,
                              void* d_out, int out_size, void* d_ws, size_t ws_size,
                              hipStream_t stream) {
  const float* x       = (const float*)d_in[0];
  const int*   ei      = (const int*)  d_in[1];
  const float* h0      = (const float*)d_in[2];
  const float* c0      = (const float*)d_in[3];
  const float* Wgat    = (const float*)d_in[4];
  const float* att_src = (const float*)d_in[5];
  const float* att_dst = (const float*)d_in[6];
  const float* bias    = (const float*)d_in[7];
  const float* Wih     = (const float*)d_in[8];
  const float* Whh     = (const float*)d_in[9];
  float* out = (float*)d_out;

  const int N = in_sizes[0] / 64;   // 100000, divisible by 16
  const int E = in_sizes[1] / 2;    // 1600000
  const size_t NH = (size_t)N * 64;
  const int Mtiles = N / 16;        // 6250

  const int* src = ei;
  const int* dst = ei + E;

  // Workspace layout (floats then bf16), all 16B-aligned by construction.
  float* ws_f   = (float*)d_ws;
  float* xs_b   = ws_f;                 // N*64
  float* agg_b  = xs_b + NH;            // N*64
  float* asrc_b = agg_b + NH;           // N
  float* adst_b = asrc_b + N;           // N
  float* m_b    = adst_b + N;           // N
  float* z_b    = m_b + N;              // N
  float* e_b    = z_b + N;              // E
  __bf16* xh_b  = (__bf16*)(e_b + E);   // N*128
  __bf16* wgt_b = xh_b + (size_t)N * 128; // 64*64
  __bf16* wct_b = wgt_b + 64 * 64;      // 256*128

  const int TB = 256;

  // 0) pack weights
  {
    int n = 64 * 64 + 256 * 128;
    prep_weights_kernel<<<(n + TB - 1) / TB, TB, 0, stream>>>(Wgat, Wih, Whh, wgt_b, wct_b);
  }
  // 1) init agg/m/z
  init_kernel<<<(int)((NH + TB - 1) / TB), TB, 0, stream>>>(agg_b, m_b, z_b, N);
  // 2) fused xs = x @ W_gat (WMMA) + attention logits
  gemm_xs_attn_kernel<<<(Mtiles * 32 + TB - 1) / TB, TB, 0, stream>>>(
      x, wgt_b, att_src, att_dst, xs_b, asrc_b, adst_b, Mtiles);
  // 3-5) scatter softmax + weighted aggregation over edges
  edge1_kernel<<<(E + TB - 1) / TB, TB, 0, stream>>>(src, dst, asrc_b, adst_b, e_b, m_b, E);
  edge2_kernel<<<(E + TB - 1) / TB, TB, 0, stream>>>(dst, e_b, m_b, z_b, E);
  {
    dim3 blk(64, 4);
    dim3 grd((E + 3) / 4);
    edge3_kernel<<<grd, blk, 0, stream>>>(src, dst, e_b, z_b, xs_b, agg_b, E);
  }
  // 6) fused A operand [tanh(agg+bias) | h0] in bf16
  build_xh_kernel<<<(int)((NH + TB - 1) / TB), TB, 0, stream>>>(agg_b, bias, h0, xh_b, N);
  // 7) fused gates GEMM (WMMA) + LSTM epilogue -> d_out (h1, h1, c1)
  gemm_lstm_kernel<<<(Mtiles * 32 + TB - 1) / TB, TB, 0, stream>>>(xh_b, wct_b, c0, out,
                                                                   Mtiles, NH);
}
